// AttentionReadout_15710990368899
// MI455X (gfx1250) — matrix operand order repaired and checked by reading, not compiled, not run on hardware
//
#include <hip/hip_runtime.h>
#include <hip/hip_bf16.h>
#include <math.h>

// ---------------------------------------------------------------------------
// AttentionReadout for MI455X (gfx1250, wave32, WMMA).
// GEMMs run as v_wmma_f32_16x16x32_bf16 (f32 accumulate); x converted to bf16
// on the fly through LDS. Segment softmax via monotone-uint atomicMax + f32
// atomicAdd; pooling uses per-tile combined atomics (batch is sorted).
// ---------------------------------------------------------------------------

typedef __attribute__((ext_vector_type(16))) __bf16 bf16x16;
typedef __attribute__((ext_vector_type(8)))  __bf16 bf16x8;
typedef __attribute__((ext_vector_type(4)))  __bf16 bf16x4;
typedef __attribute__((ext_vector_type(8)))  float  f32x8;

#define NB 1024           // number of graphs/segments
#define XS_STRIDE 264     // 256 + 8 bf16 pad (breaks LDS bank conflicts)
#define TH_STRIDE 40      // 32 + 8 bf16 pad

union BF16V { bf16x16 v; bf16x8 h[2]; };

static __device__ __forceinline__ bf16x16 make_ab(bf16x8 lo, bf16x8 hi) {
  BF16V u; u.h[0] = lo; u.h[1] = hi; return u.v;
}

static __device__ __forceinline__ f32x8 wmma_bf16(bf16x16 a, bf16x16 b, f32x8 c) {
  return __builtin_amdgcn_wmma_f32_16x16x32_bf16(false, a, false, b, (short)0, c,
                                                 false, false);
}

static __device__ __forceinline__ f32x8 zero8() {
  f32x8 z = {0.f, 0.f, 0.f, 0.f, 0.f, 0.f, 0.f, 0.f};
  return z;
}

// Order-preserving float<->uint encoding so atomicMax(u32) == float max.
static __device__ __forceinline__ unsigned enc_f32(float f) {
  unsigned u = __float_as_uint(f);
  return (u & 0x80000000u) ? ~u : (u | 0x80000000u);
}
static __device__ __forceinline__ float dec_f32(unsigned u) {
  return (u & 0x80000000u) ? __uint_as_float(u & 0x7FFFFFFFu)
                           : __uint_as_float(~u);
}

// B operand (K x 16): WT stored row-major [n][256 k] (i.e. W transposed).
// Lane: n = L&15 (column), h = L>>4 selects K half; lane reads 16 contiguous
// k at kt*32 + 16h  ->  VGPR j holds (K=16h+2j, 16h+2j+1).
static __device__ __forceinline__ bf16x16
load_b_tile(const __bf16* __restrict__ WT, int nt, int kt, int n, int h) {
  const __bf16* p = WT + (nt * 16 + n) * 256 + kt * 32 + h * 16;
  return make_ab(*(const bf16x8*)p, *(const bf16x8*)(p + 8));
}

// A operand (16 x 32) from LDS-staged x tile. Lane: m = L&15 (row), half h:
// lane reads 16 contiguous K = 16h..16h+15 (matches B).
static __device__ __forceinline__ bf16x16
load_a_xs(const __bf16* xsp, int rt, int kt, int n, int h) {
  const __bf16* p = xsp + (rt * 16 + n) * XS_STRIDE + kt * 32 + h * 16;
  return make_ab(*(const bf16x8*)p, *(const bf16x8*)(p + 8));
}

static __device__ __forceinline__ bf16x16
load_a_th(const __bf16* thp, int rt, int n, int h) {
  const __bf16* p = thp + (rt * 16 + n) * TH_STRIDE + h * 16;
  return make_ab(*(const bf16x8*)p, *(const bf16x8*)(p + 8));
}

// Stage 64 rows of x (f32 -> bf16) into padded LDS, plus segment ids.
static __device__ __forceinline__ void
stage_tile(const float* __restrict__ x, const long long* __restrict__ batch,
           __bf16* xs, int* sbatch, int R, int N, int lane) {
  for (int t = lane; t < 4096; t += 32) {   // 64 rows * 64 float4
    int row = t >> 6, c4 = t & 63;
    int node = R + row;
    float4 v = {0.f, 0.f, 0.f, 0.f};
    if (node < N) v = ((const float4*)x)[node * 64 + c4];
    bf16x4 bv;
    bv[0] = (__bf16)v.x; bv[1] = (__bf16)v.y;
    bv[2] = (__bf16)v.z; bv[3] = (__bf16)v.w;
    *(bf16x4*)(xs + row * XS_STRIDE + c4 * 4) = bv;
  }
  for (int t = lane; t < 64; t += 32) {
    int node = R + t;
    sbatch[t] = (node < N) ? (int)batch[node] : 0;
  }
}

// ---------------------------------------------------------------------------
__global__ void gap_init_kernel(float* __restrict__ out_graph,
                                unsigned* __restrict__ seg_max,
                                float* __restrict__ seg_sum) {
  int i = blockIdx.x * blockDim.x + threadIdx.x;
  if (i < NB * 256) out_graph[i] = 0.f;
  if (i < NB * 4) { seg_max[i] = 0u; seg_sum[i] = 0.f; }
}

__global__ void gap_prep_kernel(const float* __restrict__ W1,
                                const float* __restrict__ Wt,
                                const float* __restrict__ W2,
                                __bf16* __restrict__ W1T,
                                __bf16* __restrict__ WtT,
                                __bf16* __restrict__ W2T) {
  int i = blockIdx.x * blockDim.x + threadIdx.x;   // i = n*256 + k
  if (i < 256 * 256) {
    int nn = i >> 8, k = i & 255;
    W1T[i] = (__bf16)W1[k * 256 + nn];
    WtT[i] = (__bf16)Wt[k * 256 + nn];
  }
  if (i < 16 * 256) {
    int nn = i >> 8, k = i & 255;
    W2T[i] = (nn < 4) ? (__bf16)W2[k * 4 + nn] : (__bf16)0.f;
  }
}

// Phase 1: logits = tanh(x@W1 + b1) @ W2 + b2 ; atomic segment max.
__global__ void __launch_bounds__(32)
gap_logits_kernel(const float* __restrict__ x, const long long* __restrict__ batch,
                  const float* __restrict__ b1, const float* __restrict__ b2,
                  const __bf16* __restrict__ W1T, const __bf16* __restrict__ W2T,
                  float* __restrict__ ws_logits, unsigned* __restrict__ seg_max,
                  int N) {
  const int lane = threadIdx.x;
  const int R = blockIdx.x * 64;
  const int n = lane & 15, h = lane >> 4;

  __shared__ __align__(16) __bf16 xs[64 * XS_STRIDE];
  __shared__ __align__(16) __bf16 th[64 * TH_STRIDE];
  __shared__ int sbatch[64];

  stage_tile(x, batch, xs, sbatch, R, N, lane);
  __syncthreads();

  f32x8 C2[4];
#pragma unroll
  for (int rt = 0; rt < 4; ++rt) C2[rt] = zero8();

  for (int np = 0; np < 8; ++np) {            // K=32 slice of GEMM2
#pragma unroll
    for (int sub = 0; sub < 2; ++sub) {
      int nt = np * 2 + sub;                  // 16-col tile of hidden
      float bias1 = b1[nt * 16 + n];
      f32x8 C[4];
#pragma unroll
      for (int rt = 0; rt < 4; ++rt) C[rt] = zero8();
#pragma unroll
      for (int kt = 0; kt < 8; ++kt) {
        bf16x16 b = load_b_tile(W1T, nt, kt, n, h);
#pragma unroll
        for (int rt = 0; rt < 4; ++rt)
          C[rt] = wmma_bf16(load_a_xs(xs, rt, kt, n, h), b, C[rt]);
      }
      // tanh, stage as bf16 in A-tile friendly buffer (C layout: m=8h+r, col=n)
#pragma unroll
      for (int rt = 0; rt < 4; ++rt) {
#pragma unroll
        for (int r = 0; r < 8; ++r) {
          float t = tanhf(C[rt][r] + bias1);
          th[(rt * 16 + 8 * h + r) * TH_STRIDE + sub * 16 + n] = (__bf16)t;
        }
      }
    }
    __syncthreads();
    bf16x16 b2t = load_b_tile(W2T, 0, np, n, h);
#pragma unroll
    for (int rt = 0; rt < 4; ++rt)
      C2[rt] = wmma_bf16(load_a_th(th, rt, n, h), b2t, C2[rt]);
    __syncthreads();
  }

  if (n < 4) {
    float bias2 = b2[n];
#pragma unroll
    for (int rt = 0; rt < 4; ++rt) {
#pragma unroll
      for (int r = 0; r < 8; ++r) {
        int ml = rt * 16 + 8 * h + r;
        int node = R + ml;
        if (node < N) {
          float val = C2[rt][r] + bias2;
          ws_logits[node * 4 + n] = val;
          atomicMax(seg_max + (sbatch[ml] * 4 + n), enc_f32(val));
        }
      }
    }
  }
}

// Phase 2: e = exp(logit - segmax); atomic segment sum (in-place buffer).
__global__ void gap_expsum_kernel(float* __restrict__ ws_le,
                                  const long long* __restrict__ batch,
                                  const unsigned* __restrict__ seg_max,
                                  float* __restrict__ seg_sum, int N) {
  int i = blockIdx.x * blockDim.x + threadIdx.x;
  if (i < N * 4) {
    int node = i >> 2, hd = i & 3;
    int b = (int)batch[node];
    float m = dec_f32(seg_max[b * 4 + hd]);
    float e = expf(ws_le[i] - m);
    ws_le[i] = e;
    atomicAdd(seg_sum + (b * 4 + hd), e);
  }
}

// Phase 3: attn out; weighted = attn * (x@Wt + bt); segment-sum pooling.
__global__ void __launch_bounds__(32)
gap_pool_kernel(const float* __restrict__ x, const long long* __restrict__ batch,
                const float* __restrict__ bt, const __bf16* __restrict__ WtT,
                const float* __restrict__ ws_e, const float* __restrict__ seg_sum,
                float* __restrict__ out_graph, float* __restrict__ out_attn,
                int N) {
  const int lane = threadIdx.x;
  const int R = blockIdx.x * 64;
  const int n = lane & 15, h = lane >> 4;

  __shared__ __align__(16) __bf16 xs[64 * XS_STRIDE];
  __shared__ int sbatch[64];
  __shared__ float sattn[64 * 4];

  stage_tile(x, batch, xs, sbatch, R, N, lane);
  __syncthreads();

  for (int t = lane; t < 256; t += 32) {      // 64 rows * 4 heads
    int row = t >> 2, hd = t & 3;
    int node = R + row;
    float a = 0.f;
    if (node < N) {
      float e = ws_e[node * 4 + hd];
      float s = seg_sum[sbatch[row] * 4 + hd];
      a = e / (s + 1e-16f);
      out_attn[node * 4 + hd] = a;
    }
    sattn[t] = a;
  }
  __syncthreads();

  const bool full = (R + 64) <= N;

  for (int nt = 0; nt < 16; ++nt) {
    float bias = bt[nt * 16 + n];
    int head = nt >> 2;                        // HEAD_DIM = 64 = 4 col-tiles
    f32x8 C[4];
#pragma unroll
    for (int rt = 0; rt < 4; ++rt) C[rt] = zero8();
#pragma unroll
    for (int kt = 0; kt < 8; ++kt) {
      bf16x16 b = load_b_tile(WtT, nt, kt, n, h);
#pragma unroll
      for (int rt = 0; rt < 4; ++rt)
        C[rt] = wmma_bf16(load_a_xs(xs, rt, kt, n, h), b, C[rt]);
    }
#pragma unroll
    for (int rt = 0; rt < 4; ++rt) {
      int b0 = sbatch[rt * 16], b15 = sbatch[rt * 16 + 15];
      if (full && b0 == b15) {
        // whole 16-row tile in one segment (common: avg seg len ~390):
        // in-lane sum over 8 rows, combine lane halves, 1 atomic per column.
        float s = 0.f;
#pragma unroll
        for (int r = 0; r < 8; ++r) {
          int ml = rt * 16 + 8 * h + r;
          s += (C[rt][r] + bias) * sattn[ml * 4 + head];
        }
        s += __shfl_xor(s, 16, 32);
        if (h == 0) atomicAdd(out_graph + (b0 * 256 + nt * 16 + n), s);
      } else {
        // segment boundary inside tile: per-row atomics (rare).
#pragma unroll
        for (int r = 0; r < 8; ++r) {
          int ml = rt * 16 + 8 * h + r;
          int node = R + ml;
          if (node < N) {
            float v = (C[rt][r] + bias) * sattn[ml * 4 + head];
            atomicAdd(out_graph + (sbatch[ml] * 256 + nt * 16 + n), v);
          }
        }
      }
    }
  }
}

// ---------------------------------------------------------------------------
extern "C" void kernel_launch(void* const* d_in, const int* in_sizes, int n_in,
                              void* d_out, int out_size, void* d_ws, size_t ws_size,
                              hipStream_t stream) {
  const float*     x     = (const float*)d_in[0];
  const long long* batch = (const long long*)d_in[1];   // int64 in reference
  const float*     W1    = (const float*)d_in[2];
  const float*     b1    = (const float*)d_in[3];
  const float*     W2    = (const float*)d_in[4];
  const float*     b2    = (const float*)d_in[5];
  const float*     Wt    = (const float*)d_in[6];
  const float*     bt    = (const float*)d_in[7];

  const int N = in_sizes[0] / 256;

  float* out_graph = (float*)d_out;                 // [NB*256]
  float* out_attn  = (float*)d_out + NB * 256;      // [N*4]

  char* ws = (char*)d_ws;
  __bf16*   W1T     = (__bf16*)(ws + 0);            // 256*256*2 = 131072
  __bf16*   WtT     = (__bf16*)(ws + 131072);       // 131072
  __bf16*   W2T     = (__bf16*)(ws + 262144);       // 16*256*2 = 8192
  unsigned* seg_max = (unsigned*)(ws + 270336);     // NB*4*4 = 16384
  float*    seg_sum = (float*)(ws + 286720);        // 16384
  float*    ws_le   = (float*)(ws + 303104);        // N*4*4 (logits, then e)

  const int nb64 = (N + 63) / 64;

  gap_init_kernel<<<(NB * 256 + 255) / 256, 256, 0, stream>>>(out_graph, seg_max,
                                                              seg_sum);
  gap_prep_kernel<<<(256 * 256 + 255) / 256, 256, 0, stream>>>(W1, Wt, W2, W1T,
                                                               WtT, W2T);
  gap_logits_kernel<<<nb64, 32, 0, stream>>>(x, batch, b1, b2, W1T, W2T, ws_le,
                                             seg_max, N);
  gap_expsum_kernel<<<(N * 4 + 255) / 256, 256, 0, stream>>>(ws_le, batch,
                                                             seg_max, seg_sum, N);
  gap_pool_kernel<<<nb64, 32, 0, stream>>>(x, batch, bt, WtT, ws_le, seg_sum,
                                           out_graph, out_attn, N);
}
